// LSTM_Unrolled_3762391352076
// MI455X (gfx1250) — compile-verified, matrix operand order checked by reading, not run on hardware
//
#include <hip/hip_runtime.h>
#include <hip/hip_fp16.h>

typedef __attribute__((ext_vector_type(16))) _Float16 v16h;
typedef __attribute__((ext_vector_type(8)))  float    v8f;

constexpr int Bsz = 512;    // batch
constexpr int Tt  = 1024;   // timesteps
constexpr int INs = 4;      // input features
constexpr int Hh  = 64;     // hidden
constexpr int Ll  = 4;      // layers
constexpr int Cc  = 4;      // classes
constexpr int Gg  = 4 * Hh; // 256 gate columns
constexpr int BT  = 32;     // batch tile per workgroup (two WMMA M-tiles)
constexpr int WGS = 256;    // 8 waves: wave = (m_tile<<2) | j_tile
constexpr int KP  = 32;     // zero-padded K for layer-0 input GEMM
constexpr float EPSf = 1e-5f;

// f16 weight workspace layout (element offsets)
constexpr int WS_WI  = 0;                  // W_ih layers 1..3  [3][256][64]
constexpr int WS_WI0 = 3 * Gg * Hh;        // W_ih0 zero-padded [256][32]

// ---------------------------------------------------------------------------
// Fragment loaders for the 16-bit wave32 WMMA layout (ISA 7.12.2):
//   lane 0-15 -> row=lane, K groups {0-7,16-23}; lane 16-31 -> row=lane-16,
//   K groups {8-15,24-31}.
// ---------------------------------------------------------------------------
__device__ __forceinline__ v16h load_frag(const _Float16* base, int row,
                                          int stride, int kt, int halfid) {
  union { v16h v; unsigned u[8]; } f;
  const int kb = kt * 32 + halfid * 8;
  const _Float16* rp = base + row * stride;
#pragma unroll
  for (int p = 0; p < 8; ++p) {
    int k = kb + ((p < 4) ? 0 : 16) + (p & 3) * 2;
    f.u[p] = *reinterpret_cast<const unsigned*>(rp + k);
  }
  return f.v;
}

// One-time: build an f16 fragment straight from fp32 memory (cvt per element).
__device__ __forceinline__ v16h load_frag_f32(const float* base, int row,
                                              int stride, int kt, int halfid) {
  union { v16h v; _Float16 h[16]; } f;
  const int kb = kt * 32 + halfid * 8;
  const float* rp = base + row * stride;
#pragma unroll
  for (int j = 0; j < 16; ++j) {
    int k = kb + ((j < 8) ? 0 : 16) + (j & 7);
    f.h[j] = (_Float16)rp[k];
  }
  return f.v;
}

// Layer-0 x fragment: zero-padded K=32 layout means only lanes 0-15, j<4 are
// nonzero -> one global b128 load + cvt, entirely in registers (no LDS).
__device__ __forceinline__ v16h load_x_frag(const float* xrow, bool lowhalf) {
  union { v16h v; _Float16 h[16]; } f;
#pragma unroll
  for (int j = 0; j < 16; ++j) f.h[j] = (_Float16)0.0f;
  if (lowhalf) {
    float4 xv = *reinterpret_cast<const float4*>(xrow);
    f.h[0] = (_Float16)xv.x;
    f.h[1] = (_Float16)xv.y;
    f.h[2] = (_Float16)xv.z;
    f.h[3] = (_Float16)xv.w;
  }
  return f.v;
}

__device__ __forceinline__ v8f wmma_f16(v16h a, v16h b, v8f c) {
  return __builtin_amdgcn_wmma_f32_16x16x32_f16(false, a, false, b, (short)0,
                                                c, false, false);
}

// Gate activations with raw v_rcp_f32 (no IEEE division sequences).
__device__ __forceinline__ float sigmoidf_fast(float v) {
  return __builtin_amdgcn_rcpf(1.0f + __expf(-v));
}
__device__ __forceinline__ float tanhf_fast(float v) {
  return 2.0f * __builtin_amdgcn_rcpf(1.0f + __expf(-2.0f * v)) - 1.0f;
}

// Build the f16 weight workspace: Wi(1..3) and padded Wi0.
__global__ void prep_weights(const float* __restrict__ W_ih,
                             const float* __restrict__ W_ih0,
                             _Float16* __restrict__ ws) {
  int i = blockIdx.x * blockDim.x + threadIdx.x;
  if (i < 3 * Gg * Hh) ws[WS_WI + i] = (_Float16)W_ih[i];
  if (i < Gg * KP) {
    int n = i >> 5, k = i & 31;
    ws[WS_WI0 + i] = (_Float16)((k < INs) ? W_ih0[n * INs + k] : 0.0f);
  }
}

__global__ __launch_bounds__(WGS) void lstm_fused(
    const float* __restrict__ x,
    const float* __restrict__ b_ih0, const float* __restrict__ b_hh0,
    const float* __restrict__ W_hh0, const float* __restrict__ W_hh,
    const float* __restrict__ b_ih,  const float* __restrict__ b_hh,
    const float* __restrict__ fc_w,  const float* __restrict__ fc_b,
    const _Float16* __restrict__ ws,
    float* __restrict__ out) {
  // ---- LDS: 128.5 KB (CDNA5 WGP LDS is 320 KB) ----
  __shared__ __align__(16) _Float16 sWi[3][Gg * Hh];  // Wi layers 1..3: 96 KB
  __shared__ __align__(16) _Float16 sH[2][Ll][BT * Hh];  // h 2-buffer:  32 KB
  __shared__ float sLog[BT * Cc];                        // logits:     512 B

  const int tid    = threadIdx.x;
  const int wave   = tid >> 5;
  const int lane   = tid & 31;
  const int halfid = lane >> 4;
  const int ln     = lane & 15;
  const int mt     = wave >> 2;   // which 16-row M-tile
  const int jt     = wave & 3;    // which 16-col hidden tile (all 4 gates)
  const int b0     = blockIdx.x * BT;
  const int nbase  = jt * 16 + ln;  // hidden column this lane owns

  // ---- one-time staging: Wi (f16) into LDS via 16B copies ----
  {
    const uint4* src = reinterpret_cast<const uint4*>(ws + WS_WI);
    uint4* dst = reinterpret_cast<uint4*>(&sWi[0][0]);
    for (int i = tid; i < 3 * Gg * Hh / 8; i += WGS) dst[i] = src[i];
  }
  for (int idx = tid; idx < 2 * Ll * BT * Hh; idx += WGS)
    (&sH[0][0][0])[idx] = (_Float16)0.0f;

  // ---- loop-invariant per-lane biases: 16 registers, no LDS, no reloads ----
  float breg[Ll][4];
#pragma unroll
  for (int g = 0; g < 4; ++g) {
    const int n = g * 64 + nbase;
    breg[0][g] = b_ih0[n] + b_hh0[n];
    breg[1][g] = b_ih[0 * Gg + n] + b_hh[0 * Gg + n];
    breg[2][g] = b_ih[1 * Gg + n] + b_hh[1 * Gg + n];
    breg[3][g] = b_ih[2 * Gg + n] + b_hh[2 * Gg + n];
  }

  // ---- register-resident weight fragments: 36 frags = 288 VGPRs/lane ----
  // Recurrent weights for all 4 layers + padded layer-0 input weights stay
  // pinned in the 1024-deep VGPR file for the whole 1024-step recurrence.
  // Inter-layer Wi (24 more frags) streams from LDS to stay in budget.
  v16h fWi0[4];
  v16h fWh[Ll][4][2];  // [layer][gate][k-tile]
#pragma unroll
  for (int g = 0; g < 4; ++g) {
    const int n = g * 64 + nbase;
    fWi0[g] = load_frag(ws + WS_WI0, n, KP, 0, halfid);
#pragma unroll
    for (int kt = 0; kt < 2; ++kt) {
      fWh[0][g][kt] = load_frag_f32(W_hh0, n, Hh, kt, halfid);
      fWh[1][g][kt] = load_frag_f32(W_hh + 0 * Gg * Hh, n, Hh, kt, halfid);
      fWh[2][g][kt] = load_frag_f32(W_hh + 1 * Gg * Hh, n, Hh, kt, halfid);
      fWh[3][g][kt] = load_frag_f32(W_hh + 2 * Gg * Hh, n, Hh, kt, halfid);
    }
  }

  // cell state in registers: lane owns rows m = mt*16 + v + 8*halfid,
  // column j = nbase, for each layer
  float creg[Ll][8];
#pragma unroll
  for (int l = 0; l < Ll; ++l)
#pragma unroll
    for (int v = 0; v < 8; ++v) creg[l][v] = 0.0f;

  __syncthreads();

  // x rows this lane feeds into the layer-0 A fragment (lanes 0-15 only)
  const float* xBase = x + (size_t)(b0 + mt * 16 + ln) * Tt * INs;
  v16h aX = load_x_frag(xBase, halfid == 0);  // t = 0, software-pipelined

  // =============================== time loop ===============================
  for (int t = 0; t < Tt; ++t) {
    const int p = t & 1;   // h write buffer this step
    v16h aXn = aX;

#pragma unroll
    for (int l = 0; l < Ll; ++l) {
      v8f acc[4] = {};  // SRC2 starts as inline 0; bias folded in later

      if (l == 0) {
        v16h aR0 = load_frag(&sH[1 - p][0][mt * 16 * Hh], ln, Hh, 0, halfid);
        v16h aR1 = load_frag(&sH[1 - p][0][mt * 16 * Hh], ln, Hh, 1, halfid);
#pragma unroll
        for (int g = 0; g < 4; ++g) {
          acc[g] = wmma_f16(aX,  fWi0[g],      acc[g]);
          acc[g] = wmma_f16(aR0, fWh[0][g][0], acc[g]);
          acc[g] = wmma_f16(aR1, fWh[0][g][1], acc[g]);
        }
        // pipeline next step's x load behind layers 1..3 compute
        if (t + 1 < Tt)
          aXn = load_x_frag(xBase + (size_t)(t + 1) * INs, halfid == 0);
        if (halfid == 0 && t + 2 < Tt)  // global_prefetch_b8, 2 steps ahead
          __builtin_prefetch(xBase + (size_t)(t + 2) * INs, 0, 1);
      } else {
        v16h aI0 = load_frag(&sH[p][l - 1][mt * 16 * Hh], ln, Hh, 0, halfid);
        v16h aI1 = load_frag(&sH[p][l - 1][mt * 16 * Hh], ln, Hh, 1, halfid);
        v16h aR0 = load_frag(&sH[1 - p][l][mt * 16 * Hh], ln, Hh, 0, halfid);
        v16h aR1 = load_frag(&sH[1 - p][l][mt * 16 * Hh], ln, Hh, 1, halfid);
        // Wi streams from LDS.  Launder only the 32-bit offset (defeats
        // LICM hoist-and-spill), then re-establish 16B alignment so the
        // loads vectorize to ds_load_b128.
        unsigned wiOff = (unsigned)((l - 1) * Gg * Hh);
        asm volatile("" : "+s"(wiOff));
        wiOff &= ~7u;
        const _Float16* wi = (const _Float16*)__builtin_assume_aligned(
            &sWi[0][0] + wiOff, 16);
        // 2-stage software pipeline: gate g+1's B-fragments load while gate
        // g's four WMMAs execute -> DS latency fully covered.
        v16h bW0 = load_frag(wi, nbase, Hh, 0, halfid);
        v16h bW1 = load_frag(wi, nbase, Hh, 1, halfid);
#pragma unroll
        for (int g = 0; g < 4; ++g) {
          v16h nb0, nb1;
          if (g < 3) {
            nb0 = load_frag(wi, (g + 1) * 64 + nbase, Hh, 0, halfid);
            nb1 = load_frag(wi, (g + 1) * 64 + nbase, Hh, 1, halfid);
          }
          acc[g] = wmma_f16(aI0, bW0, acc[g]);
          acc[g] = wmma_f16(aI1, bW1, acc[g]);
          acc[g] = wmma_f16(aR0, fWh[l][g][0], acc[g]);
          acc[g] = wmma_f16(aR1, fWh[l][g][1], acc[g]);
          if (g < 3) {
            bW0 = nb0;
            bW1 = nb1;
          }
        }
      }

      // ---- LSTM cell: all 4 gates for (m,j) live in this lane's registers.
      _Float16* hOut = &sH[p][l][mt * 16 * Hh];
#pragma unroll
      for (int v = 0; v < 8; ++v) {
        const int m = v + 8 * halfid;  // local row within this M-tile
        float c = sigmoidf_fast(acc[1][v] + breg[l][1]) * creg[l][v] +
                  sigmoidf_fast(acc[0][v] + breg[l][0]) *
                      tanhf_fast(acc[2][v] + breg[l][2]);
        creg[l][v] = c;
        hOut[m * Hh + nbase] =
            (_Float16)(sigmoidf_fast(acc[3][v] + breg[l][3]) * tanhf_fast(c));
      }
      __syncthreads();  // publish h[l]@t before layer l+1 / next step reads it
    }

    aX = aXn;
  }

  // ---- classifier on final h of last layer + custom softmax ----
  const _Float16* hf = &sH[(Tt - 1) & 1][Ll - 1][0];
  if (tid < BT * Cc) {
    int m = tid >> 2, cls = tid & 3;
    float a = fc_b[cls];
#pragma unroll
    for (int j = 0; j < Hh; ++j)
      a += (float)hf[m * Hh + j] * fc_w[cls * Hh + j];
    sLog[tid] = __expf(a);
  }
  __syncthreads();
  if (tid < BT * Cc) {
    int m = tid >> 2, cls = tid & 3;
    float denom = sLog[m * 4 + 0] + sLog[m * 4 + 1] + sLog[m * 4 + 2] +
                  sLog[m * 4 + 3] + EPSf;
    out[(size_t)(b0 + m) * Cc + cls] = sLog[tid] / denom;
  }
}

extern "C" void kernel_launch(void* const* d_in, const int* in_sizes, int n_in,
                              void* d_out, int out_size, void* d_ws,
                              size_t ws_size, hipStream_t stream) {
  const float* x     = (const float*)d_in[0];
  const float* W_ih0 = (const float*)d_in[1];
  const float* W_hh0 = (const float*)d_in[2];
  const float* b_ih0 = (const float*)d_in[3];
  const float* b_hh0 = (const float*)d_in[4];
  const float* W_ih  = (const float*)d_in[5];
  const float* W_hh  = (const float*)d_in[6];
  const float* b_ih  = (const float*)d_in[7];
  const float* b_hh  = (const float*)d_in[8];
  const float* fc_w  = (const float*)d_in[9];
  const float* fc_b  = (const float*)d_in[10];
  float* out = (float*)d_out;

  _Float16* ws = (_Float16*)d_ws;  // 57344 halves = 112 KB
  prep_weights<<<(3 * Gg * Hh + 255) / 256, 256, 0, stream>>>(W_ih, W_ih0, ws);
  lstm_fused<<<Bsz / BT, WGS, 0, stream>>>(x, b_ih0, b_hh0, W_hh0, W_hh, b_ih,
                                           b_hh, fc_w, fc_b, ws, out);
}
